// PyramidGNN_11467562680654
// MI455X (gfx1250) — matrix operand order repaired and checked by reading, not compiled, not run on hardware
//
#include <hip/hip_runtime.h>
#include <stdint.h>

typedef __bf16 bf16;
typedef __attribute__((ext_vector_type(8)))  __bf16 v8bf;
typedef __attribute__((ext_vector_type(16))) __bf16 v16bf;
typedef __attribute__((ext_vector_type(8)))  float  v8f;

#define HEADS      4
#define D_IN       256
#define HC         1024      // HEADS * C
#define C_OUT      256
#define NEG_SLOPE  0.2f
#define NEGINF_KEY 0x007FFFFFu   // monotone-uint encoding of -inf

// ---------------------------------------------------------------- converters
__global__ void k_convert_x(const float* __restrict__ x, bf16* __restrict__ xb,
                            long long n) {
  long long i = (long long)blockIdx.x * blockDim.x + threadIdx.x;
  if (i < n) xb[i] = (bf16)x[i];
}

// W [256,1024] fp32 (row-major, K x N) -> Wt [1024,256] bf16 (N x K), so a
// WMMA B-fragment (two 8-K runs of one column, 16 apart) is two 16B loads.
__global__ void k_convert_w(const float* __restrict__ W, bf16* __restrict__ Wt) {
  int i = blockIdx.x * blockDim.x + threadIdx.x;   // i = col*256 + k
  int col = i >> 8;
  int k   = i & 255;
  Wt[i] = (bf16)W[k * HC + col];
}

// ---------------------------------------------------------------- WMMA GEMM
// h[N,1024] = xb[N,256] @ W[256,1024]   (W pre-transposed to Wt[1024,256])
// Block = 256 threads = 8 waves. Block owns 16 rows; wave w owns cols
// [128w, 128w+128) as 8 accumulators of 16x16, K-loop 8 x (16x16x32 bf16).
//
// A fragment (16x32 bf16), ISA 7.12.2: lane<16 row=lane, elems {K0..7,K16..23};
// lane>=16 row=lane-16, elems {K8..15,K24..31}.
// B fragment (32x16 bf16): lane<16 col=lane, elems {K0..7,K16..23}; lane>=16
// col=lane-16, elems {K8..15,K24..31} (two 16B runs in Wt).
__global__ __launch_bounds__(256) void k_gemm_wmma(
    const bf16* __restrict__ xb, const bf16* __restrict__ Wt,
    float* __restrict__ h) {
  const int lane    = threadIdx.x & 31;
  const int wave    = threadIdx.x >> 5;       // 0..7
  const int rowBase = blockIdx.x << 4;        // 16 rows / block
  const int half    = lane >> 4;              // 0 or 1
  const int l16     = lane & 15;

  v8f acc[8] = {};

  const bf16* aRow = xb + (size_t)(rowBase + l16) * D_IN;

  #pragma unroll
  for (int kt = 0; kt < 8; ++kt) {
    const int k0 = kt * 32;
    // ---- A fragment: two contiguous 16B runs per lane
    v8bf alo = *reinterpret_cast<const v8bf*>(aRow + k0 + half * 8);
    v8bf ahi = *reinterpret_cast<const v8bf*>(aRow + k0 + 16 + half * 8);
    v16bf A;
    #pragma unroll
    for (int i = 0; i < 8; ++i) { A[i] = alo[i]; A[i + 8] = ahi[i]; }

    const int kb = k0 + half * 8;
    #pragma unroll
    for (int nt = 0; nt < 8; ++nt) {
      const int col = wave * 128 + nt * 16 + l16;
      v8bf blo = *reinterpret_cast<const v8bf*>(Wt + (size_t)col * D_IN + kb);
      v8bf bhi = *reinterpret_cast<const v8bf*>(Wt + (size_t)col * D_IN + kb + 16);
      v16bf B;
      #pragma unroll
      for (int i = 0; i < 8; ++i) { B[i] = blo[i]; B[i + 8] = bhi[i]; }
      acc[nt] = __builtin_amdgcn_wmma_f32_16x16x32_bf16(
          false, A, false, B, (short)0, acc[nt], false, false);
    }
  }

  // C/D layout: VGPR i -> row = rowBase + half*8 + i, col = colTile + l16
  #pragma unroll
  for (int nt = 0; nt < 8; ++nt) {
    const int col = wave * 128 + nt * 16 + l16;
    float* hp = h + (size_t)(rowBase + half * 8) * HC + col;
    #pragma unroll
    for (int i = 0; i < 8; ++i) hp[(size_t)i * HC] = acc[nt][i];
  }
}

// ---------------------------------------------------------------- attention scores
// One block (4 waves) per node; wave = head; wave-reduce dot(h, a_src/a_dst).
__global__ __launch_bounds__(128) void k_scores(
    const float* __restrict__ h, const float* __restrict__ a_src,
    const float* __restrict__ a_dst, float* __restrict__ es,
    float* __restrict__ ed) {
  const int n    = blockIdx.x;
  const int head = threadIdx.x >> 5;
  const int lane = threadIdx.x & 31;

  const float4* hp = reinterpret_cast<const float4*>(
      h + (size_t)n * HC + head * C_OUT + lane * 8);
  const float4* sp = reinterpret_cast<const float4*>(a_src + head * C_OUT + lane * 8);
  const float4* dp = reinterpret_cast<const float4*>(a_dst + head * C_OUT + lane * 8);

  float ss = 0.f, sd = 0.f;
  #pragma unroll
  for (int j = 0; j < 2; ++j) {
    float4 hv = hp[j], sv = sp[j], dv = dp[j];
    ss += hv.x * sv.x + hv.y * sv.y + hv.z * sv.z + hv.w * sv.w;
    sd += hv.x * dv.x + hv.y * dv.y + hv.z * dv.z + hv.w * dv.w;
  }
  #pragma unroll
  for (int off = 16; off > 0; off >>= 1) {
    ss += __shfl_xor(ss, off, 32);
    sd += __shfl_xor(sd, off, 32);
  }
  if (lane == 0) {
    es[(size_t)n * HEADS + head] = ss;
    ed[(size_t)n * HEADS + head] = sd;
  }
}

// ---------------------------------------------------------------- init
__global__ void k_init(unsigned* __restrict__ menc, float* __restrict__ denom,
                       float* __restrict__ acc, long long n4, long long nacc) {
  long long i      = (long long)blockIdx.x * blockDim.x + threadIdx.x;
  long long stride = (long long)gridDim.x * blockDim.x;
  for (; i < nacc; i += stride) {
    acc[i] = 0.f;
    if (i < n4) { menc[i] = NEGINF_KEY; denom[i] = 0.f; }
  }
}

// ---------------------------------------------------------------- edge pass 1: leaky-relu + segment max
__global__ void k_edge_max(const float* __restrict__ es,
                           const float* __restrict__ ed,
                           const int* __restrict__ src,
                           const int* __restrict__ tgt,
                           float* __restrict__ eb,
                           unsigned* __restrict__ menc, int E) {
  int e = blockIdx.x * blockDim.x + threadIdx.x;
  if (e >= E) return;
  int s = src[e], t = tgt[e];
  float4 a = *reinterpret_cast<const float4*>(es + (size_t)s * HEADS);
  float4 b = *reinterpret_cast<const float4*>(ed + (size_t)t * HEADS);
  float v[4] = {a.x + b.x, a.y + b.y, a.z + b.z, a.w + b.w};
  #pragma unroll
  for (int hd = 0; hd < HEADS; ++hd) {
    float el = v[hd] > 0.f ? v[hd] : NEG_SLOPE * v[hd];
    v[hd] = el;
    int iv = __float_as_int(el);
    unsigned key = (iv >= 0) ? (unsigned(iv) | 0x80000000u) : ~unsigned(iv);
    atomicMax(&menc[(size_t)t * HEADS + hd], key);
  }
  *reinterpret_cast<float4*>(eb + (size_t)e * HEADS) =
      make_float4(v[0], v[1], v[2], v[3]);
}

__device__ inline float dec_key(unsigned key) {
  int iv = (key & 0x80000000u) ? int(key & 0x7FFFFFFFu) : int(~key);
  return __int_as_float(iv);
}

// ---------------------------------------------------------------- edge pass 2: exp + segment sum
__global__ void k_edge_exp(const int* __restrict__ tgt,
                           const unsigned* __restrict__ menc,
                           float* __restrict__ eb,
                           float* __restrict__ denom, int E) {
  int e = blockIdx.x * blockDim.x + threadIdx.x;
  if (e >= E) return;
  int t = tgt[e];
  float4 w4 = *reinterpret_cast<const float4*>(eb + (size_t)e * HEADS);
  float w[4] = {w4.x, w4.y, w4.z, w4.w};
  #pragma unroll
  for (int hd = 0; hd < HEADS; ++hd) {
    float m  = dec_key(menc[(size_t)t * HEADS + hd]);
    float wv = __expf(w[hd] - m);
    w[hd] = wv;
    atomicAdd(&denom[(size_t)t * HEADS + hd], wv);
  }
  *reinterpret_cast<float4*>(eb + (size_t)e * HEADS) =
      make_float4(w[0], w[1], w[2], w[3]);
}

// ---------------------------------------------------------------- edge pass 3: weighted aggregate (head mean folded as *0.25)
__global__ __launch_bounds__(256) void k_edge_aggr(
    const int* __restrict__ src, const int* __restrict__ tgt,
    const float* __restrict__ eb, const float* __restrict__ denom,
    const float* __restrict__ h, float* __restrict__ acc) {
  __shared__ float salpha[HEADS];
  const int e = blockIdx.x;
  const int s = src[e];
  const int t = tgt[e];
  if (threadIdx.x < HEADS) {
    salpha[threadIdx.x] = eb[(size_t)e * HEADS + threadIdx.x] /
                          denom[(size_t)t * HEADS + threadIdx.x] * 0.25f;
  }
  __syncthreads();
  const int c = threadIdx.x;
  const float* hb = h + (size_t)s * HC + c;
  float val = salpha[0] * hb[0]       + salpha[1] * hb[C_OUT] +
              salpha[2] * hb[2*C_OUT] + salpha[3] * hb[3*C_OUT];
  atomicAdd(&acc[(size_t)t * C_OUT + c], val);
}

// ---------------------------------------------------------------- epilogues
// layer-1: add bias and emit bf16 input of layer 2
__global__ void k_bias_tobf(const float* __restrict__ acc,
                            const float* __restrict__ b,
                            bf16* __restrict__ xb, long long n) {
  long long i = (long long)blockIdx.x * blockDim.x + threadIdx.x;
  if (i < n) xb[i] = (bf16)(acc[i] + b[i & (C_OUT - 1)]);
}
// layer-2: in-place bias on d_out
__global__ void k_bias_inplace(float* __restrict__ out,
                               const float* __restrict__ b, long long n) {
  long long i = (long long)blockIdx.x * blockDim.x + threadIdx.x;
  if (i < n) out[i] += b[i & (C_OUT - 1)];
}

// ---------------------------------------------------------------- launch
extern "C" void kernel_launch(void* const* d_in, const int* in_sizes, int n_in,
                              void* d_out, int out_size, void* d_ws, size_t ws_size,
                              hipStream_t stream) {
  const float* nodeEmb = (const float*)d_in[0];
  const float* W1      = (const float*)d_in[1];
  const float* as1     = (const float*)d_in[2];
  const float* ad1     = (const float*)d_in[3];
  const float* b1      = (const float*)d_in[4];
  const float* W2      = (const float*)d_in[5];
  const float* as2     = (const float*)d_in[6];
  const float* ad2     = (const float*)d_in[7];
  const float* b2      = (const float*)d_in[8];
  const int*   src     = (const int*)d_in[9];
  const int*   tgt     = (const int*)d_in[10];

  const long long N = (long long)in_sizes[0] / D_IN;   // 73728
  const int       E = in_sizes[9];

  // ---- workspace carve-up (256B aligned)
  uint8_t* ws  = (uint8_t*)d_ws;
  size_t   off = 0;
  auto carve = [&](size_t bytes) {
    uint8_t* p = ws + off;
    off = (off + bytes + 255) & ~(size_t)255;
    return p;
  };
  bf16*     Wt    = (bf16*)    carve((size_t)HC * D_IN * sizeof(bf16));
  bf16*     xb    = (bf16*)    carve((size_t)N * D_IN * sizeof(bf16));
  float*    h     = (float*)   carve((size_t)N * HC * sizeof(float));
  float*    es    = (float*)   carve((size_t)N * HEADS * sizeof(float));
  float*    ed    = (float*)   carve((size_t)N * HEADS * sizeof(float));
  unsigned* menc  = (unsigned*)carve((size_t)N * HEADS * sizeof(unsigned));
  float*    denom = (float*)   carve((size_t)N * HEADS * sizeof(float));
  float*    eb    = (float*)   carve((size_t)E * HEADS * sizeof(float));
  float*    xnext = (float*)   carve((size_t)N * C_OUT * sizeof(float));

  const long long nElem  = N * C_OUT;                 // N*256 elements
  const int gElem  = (int)((nElem + 255) / 256);
  const int gEdge  = (E + 255) / 256;
  const int gGemm  = (int)(N / 16);
  const int gConvW = (HC * D_IN) / 256;

  auto run_layer = [&](const float* W, const float* asv, const float* adv,
                       float* accOut) {
    k_convert_w<<<gConvW, 256, 0, stream>>>(W, Wt);
    k_gemm_wmma<<<gGemm, 256, 0, stream>>>(xb, Wt, h);
    k_scores<<<(int)N, 128, 0, stream>>>(h, asv, adv, es, ed);
    k_init<<<4096, 256, 0, stream>>>(menc, denom, accOut, N * HEADS, nElem);
    k_edge_max<<<gEdge, 256, 0, stream>>>(es, ed, src, tgt, eb, menc, E);
    k_edge_exp<<<gEdge, 256, 0, stream>>>(tgt, menc, eb, denom, E);
    k_edge_aggr<<<E, 256, 0, stream>>>(src, tgt, eb, denom, h, accOut);
  };

  // ---- layer 1
  k_convert_x<<<gElem, 256, 0, stream>>>(nodeEmb, xb, nElem);
  run_layer(W1, as1, ad1, xnext);
  k_bias_tobf<<<gElem, 256, 0, stream>>>(xnext, b1, xb, nElem);

  // ---- layer 2 (accumulate straight into d_out)
  run_layer(W2, as2, ad2, (float*)d_out);
  k_bias_inplace<<<gElem, 256, 0, stream>>>((float*)d_out, b2, nElem);
}
